// SimpleMPGNN_15152644620815
// MI455X (gfx1250) — matrix operand order repaired and checked by reading, not compiled, not run on hardware
//
#include <hip/hip_runtime.h>
#include <hip/hip_bf16.h>

// ---------------- types ----------------
typedef __attribute__((ext_vector_type(16))) __bf16        v16bf;
typedef __attribute__((ext_vector_type(8)))  float         v8f;
typedef __attribute__((ext_vector_type(4)))  unsigned int  u32x4;
typedef __attribute__((ext_vector_type(4)))  float         f32x4;
typedef __attribute__((ext_vector_type(4)))  __bf16        bf16x4;

union FragU { u32x4 q[2]; v16bf v; };

__device__ __forceinline__ v8f wmma_bf16(v16bf a, v16bf b, v8f c) {
  return __builtin_amdgcn_wmma_f32_16x16x32_bf16(false, a, false, b, (short)0, c, false, false);
}

// A fragment (16x32 bf16): lane<16 -> K 0..7 @ off 0, K 16..23 @ off 32
//                          lane>=16 -> K 8..15 @ off 16, K 24..31 @ off 48
__device__ __forceinline__ v16bf load_fragA(const __bf16* row, int kkbyte, int hi) {
  FragU f;
  const char* p = (const char*)row + kkbyte;
  f.q[0] = *(const u32x4*)(p + hi * 16);
  f.q[1] = *(const u32x4*)(p + 32 + hi * 16);
  return f.v;
}
// B fragment (32x16 bf16, stored as W^T rows): lanes 0-15 hold K 0..7, 16..23,
// lanes 16-31 hold K 8..15, 24..31 (same order as A).
__device__ __forceinline__ v16bf load_fragB(const __bf16* row, int kkbyte, int hi) {
  FragU f;
  const char* p = (const char*)row + kkbyte;
  f.q[0] = *(const u32x4*)(p + hi * 16);
  f.q[1] = *(const u32x4*)(p + 32 + hi * 16);
  return f.v;
}

__device__ __forceinline__ void atomicMaxF(float* addr, float v) {
  if (v >= 0.0f) atomicMax((int*)addr, __float_as_int(v));
  else           atomicMin((unsigned int*)addr, __float_as_uint(v));
}

// ---------------- small utility kernels ----------------
__global__ void fill_u32_kernel(unsigned int* p, unsigned int val, long long n) {
  long long i = (long long)blockIdx.x * 256 + threadIdx.x;
  if (i < n) p[i] = val;
}

// wt[n*K + k] = w[k*Nn + n]   (fp32 -> bf16, transposed)
__global__ void transpose_bf16_kernel(const float* __restrict__ w, __bf16* __restrict__ wt,
                                      int K, int Nn) {
  int i = blockIdx.x * 256 + threadIdx.x;
  if (i < K * Nn) {
    int n = i / K, k = i % K;
    wt[i] = (__bf16)w[(long long)k * Nn + n];
  }
}

// ---------------- EdgeConv: tiled bf16 WMMA ----------------
// out[dst] = max over edges of ( relu([x_i, x_j-x_i] @ W1 + b1) @ W2 + b2 )
// 256 edges per block; 8 waves; each wave owns TWO 16-edge M-strips so every
// B fragment feeds 2 WMMAs (1.25 ds_load_b128 per WMMA instead of 2.25).
template <int F>  // input feature dim; message K1 = 2F; output H = 128
__global__ __launch_bounds__(256)
void edgeconv_kernel(const float* __restrict__ x,     // [Nnodes, F]
                     const int*   __restrict__ src,
                     const int*   __restrict__ dst,
                     const __bf16* __restrict__ wt1,  // [128][2F] = W1^T
                     const __bf16* __restrict__ wt2,  // [128][128] = W2^T
                     const float* __restrict__ b1,
                     const float* __restrict__ b2,
                     float* __restrict__ out,         // [Nnodes, 128], pre-filled -inf
                     int E_total) {
  constexpr int H = 128;
  constexpr int TILE = 256;
  constexpr int K1 = 2 * F;
  constexpr int KK1 = F / 32;   // K-steps per half of GEMM1

  __shared__ __align__(16) __bf16 ldsW1[H * F];      // current half of W1^T
  __shared__ __align__(16) __bf16 ldsW2[H * H];      // W2^T
  // A-tile (TILE x F) and intermediate M-tile (TILE x H) are never live at the
  // same time -> share one buffer (H >= F).
  __shared__ __align__(16) __bf16 ldsAM[TILE * H];
  __shared__ int ldsDst[TILE];

  __bf16* ldsA = ldsAM;
  __bf16* ldsM = ldsAM;

  const int tid = threadIdx.x;
  const int wave = tid >> 5;
  const int lane = tid & 31;
  const int lane15 = lane & 15;
  const int hi = lane >> 4;
  const int e0 = blockIdx.x * TILE;
  const int m0 = wave * 32;            // first M-strip row
  const int m1 = wave * 32 + 16;       // second M-strip row

  // load W2^T once (16B vector copies)
  for (int i = tid; i < H * H / 8; i += 256)
    ((u32x4*)ldsW2)[i] = ((const u32x4*)wt2)[i];

  // GEMM1 accumulators: 2 strips x (16 edges x 128 outputs), init with bias b1
  v8f acc0[8], acc1[8];
#pragma unroll
  for (int ns = 0; ns < 8; ++ns) {
    float bv = b1[ns * 16 + lane15];
#pragma unroll
    for (int r = 0; r < 8; ++r) { acc0[ns][r] = bv; acc1[ns][r] = bv; }
  }

  // Stream the two K-halves: half 0 = x_i (dst feats), half 1 = x_j - x_i
  for (int kh = 0; kh < 2; ++kh) {
    // load this half of W1^T : ldsW1[n][k] = wt1[n*K1 + kh*F + k]
    for (int i = tid; i < H * (F / 8); i += 256) {
      int n = i / (F / 8), kq = i % (F / 8);
      ((u32x4*)ldsW1)[n * (F / 8) + kq] =
          *(const u32x4*)(wt1 + (long long)n * K1 + kh * F + kq * 8);
    }
    // build message tile half: one edge row per thread
    {
      int e = tid;
      int ge = e0 + e;
      int s = 0, d = 0;
      if (ge < E_total) { s = src[ge]; d = dst[ge]; }
      if (kh == 0) ldsDst[e] = (ge < E_total) ? d : -1;
      const float* xd = x + (long long)d * F;
      const float* xs = x + (long long)s * F;
      if (kh == 0) {
        // warm the source row for the kh=1 pass
        __builtin_prefetch(xs, 0, 0);
        __builtin_prefetch(xs + F - 1, 0, 0);
      }
#pragma unroll
      for (int k = 0; k < F; k += 4) {
        f32x4 vd = *(const f32x4*)(xd + k);
        f32x4 val;
        if (kh == 0) val = vd;
        else {
          f32x4 vs = *(const f32x4*)(xs + k);
          val = vs - vd;
        }
        bf16x4 o;
        o.x = (__bf16)val.x; o.y = (__bf16)val.y;
        o.z = (__bf16)val.z; o.w = (__bf16)val.w;
        *(bf16x4*)(ldsA + e * F + k) = o;
      }
    }
    __syncthreads();

    // accumulate this K-half
    const __bf16* aRow0 = ldsA + (m0 + lane15) * F;
    const __bf16* aRow1 = ldsA + (m1 + lane15) * F;
#pragma unroll
    for (int kk = 0; kk < KK1; ++kk) {
      v16bf fa0 = load_fragA(aRow0, kk * 64, hi);
      v16bf fa1 = load_fragA(aRow1, kk * 64, hi);
#pragma unroll
      for (int ns = 0; ns < 8; ++ns) {
        v16bf fb = load_fragB(ldsW1 + (ns * 16 + lane15) * F, kk * 64, hi);
        acc0[ns] = wmma_bf16(fa0, fb, acc0[ns]);
        acc1[ns] = wmma_bf16(fa1, fb, acc1[ns]);
      }
    }
    __syncthreads();   // before overwriting ldsW1 / ldsA (or reusing as ldsM)
  }

  // ReLU -> bf16 intermediate tile in LDS (reuses the A-tile buffer)
#pragma unroll
  for (int ns = 0; ns < 8; ++ns) {
#pragma unroll
    for (int r = 0; r < 8; ++r) {
      int nn = ns * 16 + lane15;
      float v0 = acc0[ns][r];
      float v1 = acc1[ns][r];
      ldsM[(m0 + hi * 8 + r) * H + nn] = (__bf16)(v0 > 0.0f ? v0 : 0.0f);
      ldsM[(m1 + hi * 8 + r) * H + nn] = (__bf16)(v1 > 0.0f ? v1 : 0.0f);
    }
  }
  __syncthreads();

  // GEMM2: M (256x128) @ W2 (128x128) + b2, then atomic segment-max at dst
  const __bf16* aRow0 = ldsM + (m0 + lane15) * H;
  const __bf16* aRow1 = ldsM + (m1 + lane15) * H;
#pragma unroll
  for (int ns = 0; ns < 8; ++ns) {
    v8f a20, a21;
    float bv = b2[ns * 16 + lane15];
#pragma unroll
    for (int r = 0; r < 8; ++r) { a20[r] = bv; a21[r] = bv; }
#pragma unroll
    for (int kk = 0; kk < 4; ++kk) {
      v16bf fa0 = load_fragA(aRow0, kk * 64, hi);
      v16bf fa1 = load_fragA(aRow1, kk * 64, hi);
      v16bf fb  = load_fragB(ldsW2 + (ns * 16 + lane15) * H, kk * 64, hi);
      a20 = wmma_bf16(fa0, fb, a20);
      a21 = wmma_bf16(fa1, fb, a21);
    }
#pragma unroll
    for (int r = 0; r < 8; ++r) {
      int nn = ns * 16 + lane15;
      int node0 = ldsDst[m0 + hi * 8 + r];
      int node1 = ldsDst[m1 + hi * 8 + r];
      if (node0 >= 0) atomicMaxF(out + (long long)node0 * H + nn, a20[r]);
      if (node1 >= 0) atomicMaxF(out + (long long)node1 * H + nn, a21[r]);
    }
  }
}

// ---------------- BN stats: sanitize (-inf -> 0) + column sum/sumsq ----------------
__global__ void bn_stats_kernel(float* __restrict__ buf, float* __restrict__ sums,
                                float* __restrict__ sumsq, int Nrows) {
  int col = threadIdx.x & 127;
  int part = threadIdx.x >> 7;
  int r0 = blockIdx.x * 64 + part * 32;
  float s = 0.0f, s2 = 0.0f;
  for (int r = r0; r < r0 + 32 && r < Nrows; ++r) {
    long long idx = (long long)r * 128 + col;
    float v = buf[idx];
    if (!__builtin_isfinite(v)) { v = 0.0f; buf[idx] = 0.0f; }
    s += v; s2 += v * v;
  }
  atomicAdd(&sums[col], s);
  atomicAdd(&sumsq[col], s2);
}

__global__ void bn_apply_kernel(const float* __restrict__ buf, const float* __restrict__ sums,
                                const float* __restrict__ sumsq, const float* __restrict__ g,
                                const float* __restrict__ b, float* __restrict__ outp,
                                int Nrows) {
  long long i = (long long)blockIdx.x * 256 + threadIdx.x;
  if (i >= (long long)Nrows * 128) return;
  int col = (int)(i & 127);
  float inv = 1.0f / (float)Nrows;
  float mu = sums[col] * inv;
  float var = sumsq[col] * inv - mu * mu;
  float v = (buf[i] - mu) * rsqrtf(var + 1e-5f) * g[col] + b[col];
  outp[i] = fmaxf(v, 0.0f);
}

// ---------------- global mean pooling ----------------
__global__ void pool_sum_kernel(const float* __restrict__ h, const int* __restrict__ batch,
                                float* __restrict__ gsum, float* __restrict__ cnt, int Nrows) {
  long long i = (long long)blockIdx.x * 256 + threadIdx.x;
  if (i >= (long long)Nrows * 128) return;
  int row = (int)(i >> 7), col = (int)(i & 127);
  int bI = batch[row];
  atomicAdd(&gsum[bI * 128 + col], h[i]);
  if (col == 0) atomicAdd(&cnt[bI], 1.0f);
}

// ---------------- head MLP: [64,128] -> 256 -> 128 -> 16 (fp32, single block) ----------------
__global__ __launch_bounds__(256)
void head_mlp_kernel(const float* __restrict__ gsum, const float* __restrict__ cnt,
                     const float* __restrict__ w1, const float* __restrict__ b1,
                     const float* __restrict__ w2, const float* __restrict__ b2,
                     const float* __restrict__ w3, const float* __restrict__ b3,
                     float* __restrict__ outp) {
  __shared__ float G[64 * 128];
  __shared__ float G1[64 * 256];
  __shared__ float G2[64 * 128];
  int t = threadIdx.x;
  for (int i = t; i < 64 * 128; i += 256) {
    int r = i >> 7;
    G[i] = gsum[i] / fmaxf(cnt[r], 1.0f);
  }
  __syncthreads();
  for (int i = t; i < 64 * 256; i += 256) {
    int r = i >> 8, c = i & 255;
    float a = b1[c];
    for (int k = 0; k < 128; ++k) a += G[r * 128 + k] * w1[k * 256 + c];
    G1[i] = fmaxf(a, 0.0f);
  }
  __syncthreads();
  for (int i = t; i < 64 * 128; i += 256) {
    int r = i >> 7, c = i & 127;
    float a = b2[c];
    for (int k = 0; k < 256; ++k) a += G1[r * 256 + k] * w2[k * 128 + c];
    G2[i] = fmaxf(a, 0.0f);
  }
  __syncthreads();
  for (int i = t; i < 64 * 16; i += 256) {
    int r = i >> 4, c = i & 15;
    float a = b3[c];
    for (int k = 0; k < 128; ++k) a += G2[r * 128 + k] * w3[k * 16 + c];
    outp[i] = fmaxf(a, 0.0f);
  }
}

// ---------------- launch ----------------
extern "C" void kernel_launch(void* const* d_in, const int* in_sizes, int n_in,
                              void* d_out, int out_size, void* d_ws, size_t ws_size,
                              hipStream_t stream) {
  const float* x    = (const float*)d_in[0];
  const int*   ei   = (const int*)d_in[1];
  const int*   batch= (const int*)d_in[2];
  const float* c1w1 = (const float*)d_in[3];
  const float* c1b1 = (const float*)d_in[4];
  const float* c1w2 = (const float*)d_in[5];
  const float* c1b2 = (const float*)d_in[6];
  const float* bn1g = (const float*)d_in[7];
  const float* bn1b = (const float*)d_in[8];
  const float* c2w1 = (const float*)d_in[9];
  const float* c2b1 = (const float*)d_in[10];
  const float* c2w2 = (const float*)d_in[11];
  const float* c2b2 = (const float*)d_in[12];
  const float* bn2g = (const float*)d_in[13];
  const float* bn2b = (const float*)d_in[14];
  const float* l1w  = (const float*)d_in[15];
  const float* l1b  = (const float*)d_in[16];
  const float* l2w  = (const float*)d_in[17];
  const float* l2b  = (const float*)d_in[18];
  const float* ow   = (const float*)d_in[19];
  const float* ob   = (const float*)d_in[20];

  const int E = in_sizes[1] / 2;
  const int N = in_sizes[2];
  const int* src = ei;
  const int* dst = ei + E;
  const int H = 128;

  // workspace layout
  char* ws = (char*)d_ws;
  size_t off = 0;
  auto walloc = [&](size_t bytes) -> char* {
    char* p = ws + off;
    off += (bytes + 255) & ~(size_t)255;
    return p;
  };
  float*  bufA  = (float*)walloc((size_t)N * H * 4);
  float*  bufB  = (float*)walloc((size_t)N * H * 4);
  __bf16* wt1a  = (__bf16*)walloc(128 * 128 * 2);
  __bf16* wt2a  = (__bf16*)walloc(128 * 128 * 2);
  __bf16* wt1b  = (__bf16*)walloc(128 * 256 * 2);
  __bf16* wt2b  = (__bf16*)walloc(128 * 128 * 2);
  float*  sums  = (float*)walloc(128 * 4);
  float*  sumsq = (float*)walloc(128 * 4);
  float*  gsum  = (float*)walloc(64 * 128 * 4 + 64 * 4);  // gsum then cnt contiguous
  float*  cnt   = gsum + 64 * 128;

  const long long nh = (long long)N * H;
  const int gridNH = (int)((nh + 255) / 256);
  const int gridE  = (E + 255) / 256;

  // 1) weight prep: W^T in bf16
  transpose_bf16_kernel<<<(128 * 128 + 255) / 256, 256, 0, stream>>>(c1w1, wt1a, 128, 128);
  transpose_bf16_kernel<<<(128 * 128 + 255) / 256, 256, 0, stream>>>(c1w2, wt2a, 128, 128);
  transpose_bf16_kernel<<<(256 * 128 + 255) / 256, 256, 0, stream>>>(c2w1, wt1b, 256, 128);
  transpose_bf16_kernel<<<(128 * 128 + 255) / 256, 256, 0, stream>>>(c2w2, wt2b, 128, 128);

  // 2) conv1: out -> bufA (init -inf for segment max)
  fill_u32_kernel<<<gridNH, 256, 0, stream>>>((unsigned int*)bufA, 0xFF800000u, nh);
  edgeconv_kernel<64><<<gridE, 256, 0, stream>>>(x, src, dst, wt1a, wt2a, c1b1, c1b2, bufA, E);

  // 3) BN1 + ReLU -> bufB
  fill_u32_kernel<<<1, 256, 0, stream>>>((unsigned int*)sums, 0u, 256);  // sums+sumsq contiguous
  bn_stats_kernel<<<(N + 63) / 64, 256, 0, stream>>>(bufA, sums, sumsq, N);
  bn_apply_kernel<<<gridNH, 256, 0, stream>>>(bufA, sums, sumsq, bn1g, bn1b, bufB, N);

  // 4) conv2: reads bufB, out -> bufA
  fill_u32_kernel<<<gridNH, 256, 0, stream>>>((unsigned int*)bufA, 0xFF800000u, nh);
  edgeconv_kernel<128><<<gridE, 256, 0, stream>>>(bufB, src, dst, wt1b, wt2b, c2b1, c2b2, bufA, E);

  // 5) BN2 + ReLU -> bufB
  fill_u32_kernel<<<1, 256, 0, stream>>>((unsigned int*)sums, 0u, 256);
  bn_stats_kernel<<<(N + 63) / 64, 256, 0, stream>>>(bufA, sums, sumsq, N);
  bn_apply_kernel<<<gridNH, 256, 0, stream>>>(bufA, sums, sumsq, bn2g, bn2b, bufB, N);

  // 6) global mean pooling
  fill_u32_kernel<<<(64 * 128 + 64 + 255) / 256, 256, 0, stream>>>((unsigned int*)gsum, 0u,
                                                                   64 * 128 + 64);
  pool_sum_kernel<<<gridNH, 256, 0, stream>>>(bufB, batch, gsum, cnt, N);

  // 7) head MLP -> d_out [64,16]
  head_mlp_kernel<<<1, 256, 0, stream>>>(gsum, cnt, l1w, l1b, l2w, l2b, ow, ob, (float*)d_out);
}
